// DisentangledSelfAttention_62766652064342
// MI455X (gfx1250) — matrix operand order repaired and checked by reading, not compiled, not run on hardware
//
#include <hip/hip_runtime.h>
#include <hip/hip_bf16.h>

typedef __attribute__((ext_vector_type(16))) __bf16 v16bf;
typedef __attribute__((ext_vector_type(8)))  float  v8f;

#define WMMA_BF16(a, b, c) \
  __builtin_amdgcn_wmma_f32_16x16x32_bf16(false, (a), false, (b), (short)0, (c), false, false)

static __device__ __forceinline__ v8f vzero8() {
  v8f z = {0.f, 0.f, 0.f, 0.f, 0.f, 0.f, 0.f, 0.f};
  return z;
}

// A-fragment (16x32, bf16), row-major source with leading dim `ld`.
// Lane L: m = row0 + (L&15); half = L>>4.
// elems 0..7  -> K = k0 + half*16 + j
// elems 8..15 -> K = k0 + half*16 + 8 + j
static __device__ __forceinline__ v16bf load_a_frag(const __bf16* base, int ld,
                                                    int row0, int k0, int lane) {
  const int m = row0 + (lane & 15);
  const int half = (lane >> 4) & 1;
  const __bf16* p0 = base + m * ld + k0 + half * 16;
  const __bf16* p1 = base + m * ld + k0 + half * 16 + 8;
  v16bf a;
#pragma unroll
  for (int j = 0; j < 8; ++j) a[j] = p0[j];
#pragma unroll
  for (int j = 0; j < 8; ++j) a[8 + j] = p1[j];
  return a;
}

// B-fragment (32x16) where B = M^T for row-major M (rows n, cols k, ld given).
// Lane L: n = n0 + (L&15); half = L>>4; elem j -> K = k0 + half*16 + j.
static __device__ __forceinline__ v16bf load_bT_frag(const __bf16* base, int ld,
                                                     int n0, int k0, int lane) {
  const int n = n0 + (lane & 15);
  const int half = (lane >> 4) & 1;
  const __bf16* p = base + n * ld + k0 + half * 16;
  v16bf b;
#pragma unroll
  for (int j = 0; j < 16; ++j) b[j] = p[j];
  return b;
}

// B-fragment from a clamped positional window: row index = clamp(p0 + n, 0, 1023)
static __device__ __forceinline__ v16bf load_pos_frag(const __bf16* P, int p0,
                                                      int w0, int k0, int lane) {
  int w = w0 + (lane & 15);
  const int half = (lane >> 4) & 1;
  int pidx = p0 + w;
  pidx = pidx < 0 ? 0 : (pidx > 1023 ? 1023 : pidx);
  const __bf16* p = P + pidx * 64 + k0 + half * 16;
  v16bf b;
#pragma unroll
  for (int j = 0; j < 16; ++j) b[j] = p[j];
  return b;
}

// B-fragment of a non-transposed 64x64 LDS matrix (K-dim = rows): column reads.
static __device__ __forceinline__ v16bf load_vcol_frag(const __bf16* Vl, int n0,
                                                       int k0, int lane) {
  const int n = n0 + (lane & 15);
  const int half = (lane >> 4) & 1;
  v16bf b;
#pragma unroll
  for (int j = 0; j < 16; ++j) b[j] = Vl[(k0 + half * 16 + j) * 64 + n];
  return b;
}

// ---------------------------------------------------------------------------
// Projection GEMM: dst[b,h,s,d] (bf16) = (A[M,1024] @ W[1024,1024]^T + bias)*scale
// grid: (Ncols/64, Mrows/64), block 256 threads (8 waves).
// ---------------------------------------------------------------------------
__global__ __launch_bounds__(256) void proj_gemm_kernel(
    const float* __restrict__ A, const float* __restrict__ W,
    const float* __restrict__ bias, float scale, __bf16* __restrict__ dst) {
  __shared__ __bf16 sA[64 * 32];
  __shared__ __bf16 sW[64 * 32];

  const int t = threadIdx.x;
  const int lane = t & 31, wid = t >> 5;
  const int r0 = 16 * (wid & 3);
  const int c0 = 32 * (wid >> 2);
  const int Mblk = blockIdx.y * 64;
  const int Nblk = blockIdx.x * 64;

  v8f acc0 = vzero8();
  v8f acc1 = vzero8();

  const int lr = t >> 2;        // load row 0..63
  const int lc = (t & 3) * 8;   // load col 0,8,16,24

  for (int kb = 0; kb < 1024; kb += 32) {
    __syncthreads();
    const float* ap = A + (size_t)(Mblk + lr) * 1024 + kb + lc;
    const float* wp = W + (size_t)(Nblk + lr) * 1024 + kb + lc;
#pragma unroll
    for (int j = 0; j < 8; ++j) {
      sA[lr * 32 + lc + j] = (__bf16)ap[j];
      sW[lr * 32 + lc + j] = (__bf16)wp[j];
    }
    __syncthreads();
    v16bf a  = load_a_frag(sA, 32, r0, 0, lane);
    v16bf b0 = load_bT_frag(sW, 32, c0, 0, lane);
    v16bf b1 = load_bT_frag(sW, 32, c0 + 16, 0, lane);
    acc0 = WMMA_BF16(a, b0, acc0);
    acc1 = WMMA_BF16(a, b1, acc1);
  }

  const int half = lane >> 4;
  const int nl = lane & 15;
#pragma unroll
  for (int f = 0; f < 2; ++f) {
    v8f acc = f ? acc1 : acc0;
    const int gcol = Nblk + c0 + 16 * f + nl;
    const float bv = bias ? bias[gcol] : 0.0f;
    const int hcol = gcol >> 6, dcol = gcol & 63;
#pragma unroll
    for (int k = 0; k < 8; ++k) {
      const int mg = Mblk + r0 + k + 8 * half;
      const int bidx = mg >> 10, s = mg & 1023;
      const float v = (acc[k] + bv) * scale;
      dst[(((size_t)bidx * 16 + hcol) * 1024 + s) * 64 + dcol] = (__bf16)v;
    }
  }
}

// ---------------------------------------------------------------------------
// Disentangled flash attention.
// grid: (S/64, B*h); block 256 threads (8 waves).
// Each WG: one (b,h), 64 query rows; loops 16 key tiles of 64.
// V tiles are staged with CDNA5 async global->LDS copies (ASYNCcnt).
// ---------------------------------------------------------------------------
__global__ __launch_bounds__(256) void attn_kernel(
    const __bf16* __restrict__ Q, const __bf16* __restrict__ K,
    const __bf16* __restrict__ V, const __bf16* __restrict__ PK,
    const __bf16* __restrict__ PQ, float* __restrict__ out) {
  __shared__ float  sS[64 * 64];     // scores f32; reused as bf16 P matrix
  __shared__ __align__(16) __bf16 sV[64 * 64];
  __shared__ __bf16 sA2P[64 * 128];
  __shared__ __bf16 sP2C[64 * 128];
  __shared__ float  sRed[64 * 4];
  __shared__ float  sAlpha[64];
  __shared__ float  sLinv[64];
  __bf16* sP = (__bf16*)sS;

  const int t = threadIdx.x, lane = t & 31, wid = t >> 5;
  const int r0 = 16 * (wid & 3);
  const int cgrp = wid >> 2;
  const int c0 = 32 * cgrp;
  const int i0 = blockIdx.x * 64;
  const int bh = blockIdx.y;
  const int b = bh >> 4, h = bh & 15;
  const int half = lane >> 4, nl = lane & 15;

  const __bf16* Qb  = Q + (size_t)bh * 1024 * 64;
  const __bf16* Kb  = K + (size_t)bh * 1024 * 64;
  const __bf16* Vb  = V + (size_t)bh * 1024 * 64;
  const __bf16* PKh = PK + (size_t)h * 1024 * 64;
  const __bf16* PQh = PQ + (size_t)h * 1024 * 64;

  v8f O0 = vzero8(), O1 = vzero8();
  float m_run = -1e30f, l_run = 0.0f;
  const int prow = t >> 2, pslot = t & 3;

  // Q fragments are k-tile invariant: hoist.
  const v16bf aQ0 = load_a_frag(Qb + (size_t)i0 * 64, 64, r0, 0, lane);
  const v16bf aQ1 = load_a_frag(Qb + (size_t)i0 * 64, 64, r0, 32, lane);

  for (int j0 = 0; j0 < 1024; j0 += 64) {
    __syncthreads();
    // Stage V tile via async global->LDS DMA (32 bytes per thread, two B128).
    // INST_OFFSET is added to both the LDS and global address (ISA 15.18.3),
    // so the same base registers serve both issues.
    {
      const int vr = t >> 2;          // 0..63
      const int vc = (t & 3) * 16;    // 0,16,32,48 (bf16 elements; 32B span)
      const __bf16* vp = Vb + (size_t)(j0 + vr) * 64 + vc;
      unsigned lds0 = (unsigned)(size_t)(void*)&sV[vr * 64 + vc];
      unsigned long long g0 = (unsigned long long)(size_t)(const void*)vp;
      asm volatile("global_load_async_to_lds_b128 %0, %1, off"
                   :: "v"(lds0), "v"(g0) : "memory");
      asm volatile("global_load_async_to_lds_b128 %0, %1, off offset:16"
                   :: "v"(lds0), "v"(g0) : "memory");
    }
    const v16bf aK0 = load_a_frag(Kb + (size_t)j0 * 64, 64, r0, 0, lane);
    const v16bf aK1 = load_a_frag(Kb + (size_t)j0 * 64, 64, r0, 32, lane);
    const int p0 = i0 - j0 + 449;  // window base: c(i-j) - t, t = li-lj+63

    // A2P = Q * PKwin^T (64x128); P2C = K * PQwin^T (64x128)
#pragma unroll
    for (int f = 0; f < 4; ++f) {
      const int w0 = cgrp * 64 + 16 * f;
      {
        v16bf b0 = load_pos_frag(PKh, p0, w0, 0, lane);
        v16bf b1 = load_pos_frag(PKh, p0, w0, 32, lane);
        v8f acc = vzero8();
        acc = WMMA_BF16(aQ0, b0, acc);
        acc = WMMA_BF16(aQ1, b1, acc);
#pragma unroll
        for (int k = 0; k < 8; ++k)
          sA2P[(r0 + k + 8 * half) * 128 + w0 + nl] = (__bf16)acc[k];
      }
      {
        v16bf b0 = load_pos_frag(PQh, p0, w0, 0, lane);
        v16bf b1 = load_pos_frag(PQh, p0, w0, 32, lane);
        v8f acc = vzero8();
        acc = WMMA_BF16(aK0, b0, acc);
        acc = WMMA_BF16(aK1, b1, acc);
#pragma unroll
        for (int k = 0; k < 8; ++k)
          sP2C[(r0 + k + 8 * half) * 128 + w0 + nl] = (__bf16)acc[k];
      }
    }
    __syncthreads();

    // S = Q*K^T with fused positional gather-add; write combined scores
#pragma unroll
    for (int f = 0; f < 2; ++f) {
      const int n0 = c0 + 16 * f;
      v16bf b0 = load_bT_frag(Kb + (size_t)j0 * 64, 64, n0, 0, lane);
      v16bf b1 = load_bT_frag(Kb + (size_t)j0 * 64, 64, n0, 32, lane);
      v8f acc = vzero8();
      acc = WMMA_BF16(aQ0, b0, acc);
      acc = WMMA_BF16(aQ1, b1, acc);
      const int lj = n0 + nl;
#pragma unroll
      for (int k = 0; k < 8; ++k) {
        const int li = r0 + k + 8 * half;
        const int tt = li - lj + 63;  // in [0,126]
        const float v = acc[k] + (float)sA2P[li * 128 + tt] + (float)sP2C[lj * 128 + tt];
        sS[li * 64 + lj] = v;
      }
    }
    __syncthreads();

    // Online softmax (thread-mapped: 4 threads x 16 cols per row)
    float sv[16];
#pragma unroll
    for (int e = 0; e < 16; ++e) sv[e] = sS[prow * 64 + pslot * 16 + e];
    float mloc = sv[0];
#pragma unroll
    for (int e = 1; e < 16; ++e) mloc = fmaxf(mloc, sv[e]);
    sRed[prow * 4 + pslot] = mloc;
    __syncthreads();
    float mnew = m_run;
#pragma unroll
    for (int s2 = 0; s2 < 4; ++s2) mnew = fmaxf(mnew, sRed[prow * 4 + s2]);
    const float alpha = __expf(m_run - mnew);
    float psum = 0.0f, pv[16];
#pragma unroll
    for (int e = 0; e < 16; ++e) {
      pv[e] = __expf(sv[e] - mnew);
      psum += pv[e];
    }
    __syncthreads();
    sRed[prow * 4 + pslot] = psum;
    __syncthreads();
    float rsum = 0.0f;
#pragma unroll
    for (int s2 = 0; s2 < 4; ++s2) rsum += sRed[prow * 4 + s2];
    l_run = l_run * alpha + rsum;
    m_run = mnew;
    if (pslot == 0) sAlpha[prow] = alpha;
#pragma unroll
    for (int e = 0; e < 16; ++e)
      sP[prow * 64 + pslot * 16 + e] = (__bf16)pv[e];

    // Ensure this thread's async V-tile writes have landed in LDS, then make
    // them (and P/alpha) visible workgroup-wide before the P*V WMMAs.
    asm volatile("s_wait_asynccnt 0" ::: "memory");
    __syncthreads();

    // O = diag(alpha)*O + P*V  (WMMA)
#pragma unroll
    for (int k = 0; k < 8; ++k) {
      const float a = sAlpha[r0 + k + 8 * half];
      O0[k] *= a;
      O1[k] *= a;
    }
    v16bf aP0 = load_a_frag(sP, 64, r0, 0, lane);
    v16bf aP1 = load_a_frag(sP, 64, r0, 32, lane);
    v16bf bV00 = load_vcol_frag(sV, c0, 0, lane);
    v16bf bV01 = load_vcol_frag(sV, c0, 32, lane);
    v16bf bV10 = load_vcol_frag(sV, c0 + 16, 0, lane);
    v16bf bV11 = load_vcol_frag(sV, c0 + 16, 32, lane);
    O0 = WMMA_BF16(aP0, bV00, O0);
    O0 = WMMA_BF16(aP1, bV01, O0);
    O1 = WMMA_BF16(aP0, bV10, O1);
    O1 = WMMA_BF16(aP1, bV11, O1);
  }

  if (pslot == 0) sLinv[prow] = (l_run > 0.0f) ? (1.0f / l_run) : 0.0f;
  __syncthreads();

#pragma unroll
  for (int f = 0; f < 2; ++f) {
    v8f O = f ? O1 : O0;
    const int dcol = c0 + 16 * f + nl;
#pragma unroll
    for (int k = 0; k < 8; ++k) {
      const int li = r0 + k + 8 * half;
      out[((size_t)b * 1024 + i0 + li) * 1024 + h * 64 + dcol] = O[k] * sLinv[li];
    }
  }
}

// ---------------------------------------------------------------------------
extern "C" void kernel_launch(void* const* d_in, const int* in_sizes, int n_in,
                              void* d_out, int out_size, void* d_ws, size_t ws_size,
                              hipStream_t stream) {
  (void)in_sizes; (void)n_in; (void)out_size; (void)ws_size;
  const float* hidden = (const float*)d_in[0];
  const float* rel    = (const float*)d_in[1];
  const float* Wq     = (const float*)d_in[2];
  const float* bq     = (const float*)d_in[3];
  const float* Wk     = (const float*)d_in[4];
  const float* bk     = (const float*)d_in[5];
  const float* Wv     = (const float*)d_in[6];
  const float* bv     = (const float*)d_in[7];
  const float* Wpos   = (const float*)d_in[8];
  const float* Wposq  = (const float*)d_in[9];
  const float* bposq  = (const float*)d_in[10];
  // d_in[11] attention_mask: all-true in this problem -> no-op, ignored.

  char* ws = (char*)d_ws;
  __bf16* Qbf  = (__bf16*)(ws);                       // 4 MB [b,h,s,d]
  __bf16* Kbf  = (__bf16*)(ws + (size_t)(4u << 20));  // 4 MB
  __bf16* Vbf  = (__bf16*)(ws + (size_t)(8u << 20));  // 4 MB
  __bf16* PKbf = (__bf16*)(ws + (size_t)(12u << 20)); // 2 MB [h,p,d]
  __bf16* PQbf = (__bf16*)(ws + (size_t)(14u << 20)); // 2 MB

  const float invs = 0.07216878364870323f;  // 1/sqrt(64*3)
  dim3 blk(256);

  proj_gemm_kernel<<<dim3(16, 32), blk, 0, stream>>>(hidden, Wq, bq, invs, Qbf);
  proj_gemm_kernel<<<dim3(16, 32), blk, 0, stream>>>(hidden, Wk, bk, 1.0f, Kbf);
  proj_gemm_kernel<<<dim3(16, 32), blk, 0, stream>>>(hidden, Wv, bv, 1.0f, Vbf);
  proj_gemm_kernel<<<dim3(16, 16), blk, 0, stream>>>(rel, Wpos, nullptr, 1.0f, PKbf);
  proj_gemm_kernel<<<dim3(16, 16), blk, 0, stream>>>(rel, Wposq, bposq, invs, PQbf);

  attn_kernel<<<dim3(16, 32), blk, 0, stream>>>(Qbf, Kbf, Vbf, PKbf, PQbf,
                                                (float*)d_out);
}
